// DSVTMiddleEncoder_20091857011186
// MI455X (gfx1250) — matrix operand order repaired and checked by reading, not compiled, not run on hardware
//
#include <hip/hip_runtime.h>
#include <hip/hip_bf16.h>
#include <math.h>

// Problem constants (from reference)
#define C_DIM   192
#define H_DIM   8
#define DH      24
#define SETS    2048
#define KSET    36
#define NVOX    (SETS * KSET)   // 73728
#define FDIM    384
#define LAYERS  8
#define NBLOCKS 4

typedef __bf16        v16bf  __attribute__((ext_vector_type(16)));
typedef float         v8f    __attribute__((ext_vector_type(8)));
typedef unsigned int  uint4v __attribute__((ext_vector_type(4)));

union FragBF { v16bf v; uint4v u[2]; };

// ---------------------------------------------------------------------------
// fp32 -> bf16 elementwise convert (weights)
// ---------------------------------------------------------------------------
__global__ __launch_bounds__(256) void f32_to_bf16_kernel(
    const float* __restrict__ in, __hip_bfloat16* __restrict__ out, size_t n) {
  size_t i = (size_t)blockIdx.x * blockDim.x + threadIdx.x;
  if (i < n) out[i] = __float2bfloat16(in[i]);
}

// ---------------------------------------------------------------------------
// Gather voxels into set order, add pos embed, emit bf16 A-matrices:
//   g[r,c]     = x[inds[r], c]
//   qk_in[r,c] = x[inds[r], c] + pos[inds[r], c]
// ---------------------------------------------------------------------------
__global__ __launch_bounds__(256) void gather_prep_kernel(
    const float* __restrict__ x, const float* __restrict__ pos,
    const int* __restrict__ inds,
    __hip_bfloat16* __restrict__ qk_in, __hip_bfloat16* __restrict__ g) {
  size_t i = (size_t)blockIdx.x * blockDim.x + threadIdx.x;
  if (i >= (size_t)NVOX * C_DIM) return;
  int r = (int)(i / C_DIM);
  int c = (int)(i % C_DIM);
  size_t src = (size_t)inds[r] * C_DIM + c;
  float gv = x[src];
  g[i] = __float2bfloat16(gv);
  qk_in[i] = __float2bfloat16(gv + pos[src]);
}

// ---------------------------------------------------------------------------
// Register-blocked WMMA GEMM:  out[m,n] = act( A[m,:] . W[n,:] + bias[n] )
//   A: (M x Kd) bf16 row-major; W: (Ncols x Kd) bf16 row-major (A @ W^T)
// One wave owns a 32x64 output tile: 2 A-frags x 4 B-frags -> 8 WMMAs per
// K-step of 32, fed by 12 x b128 loads (1.5 loads/WMMA). Fragment loads
// use contiguous K = half*16 + j, same order in A and B.
// Epilogue: optional exact GELU, optional row scatter, fp32 or bf16 output.
// Requires M % 32 == 0, Ncols % 64 == 0, Kd % 32 == 0 (true for all calls).
// ---------------------------------------------------------------------------
__global__ __launch_bounds__(256) void gemm_bf16_wmma_kernel(
    const __hip_bfloat16* __restrict__ A, int lda,
    const __hip_bfloat16* __restrict__ W, int ldw,
    const float* __restrict__ bias,
    int M, int Ncols, int Kd,
    const int* __restrict__ scatter,
    float* __restrict__ outF, __hip_bfloat16* __restrict__ outB, int ldo,
    int act) {
  int lane = threadIdx.x & 31;
  int wid  = threadIdx.x >> 5;
  int tilesN = Ncols >> 6;            // 64-wide N tiles
  int tilesM = M >> 5;                // 32-tall M tiles
  int gid = blockIdx.x * 8 + wid;
  if (gid >= tilesM * tilesN) return;
  int mT = gid / tilesN, nT = gid % tilesN;
  int m0 = mT << 5, n0 = nT << 6;
  int hl  = lane >> 4;                // half-wave: 0 or 1
  int l16 = lane & 15;

  // A: lanes 0..31 all map M = m0 + mi*16 + (lane&15); half selects K block.
  const __hip_bfloat16* Ap0 = A + (size_t)(m0 + l16) * lda;
  const __hip_bfloat16* Ap1 = Ap0 + (size_t)16 * lda;
  // B: N = n0 + ni*16 + (lane&15); half selects K=0..15 vs 16..31.
  const __hip_bfloat16* Wp0 = W + (size_t)(n0 + l16) * ldw;
  const __hip_bfloat16* Wp1 = Wp0 + (size_t)16 * ldw;
  const __hip_bfloat16* Wp2 = Wp0 + (size_t)32 * ldw;
  const __hip_bfloat16* Wp3 = Wp0 + (size_t)48 * ldw;

  v8f acc[2][4] = {};
  int nk = Kd >> 5;
  for (int kt = 0; kt < nk; ++kt) {
    int k0 = kt << 5;
    FragBF a[2], b[4];
    // A 16x32 bf16: elem j -> K = k0 + half*16 + j
    a[0].u[0] = *reinterpret_cast<const uint4v*>(Ap0 + k0 + hl * 16);
    a[0].u[1] = *reinterpret_cast<const uint4v*>(Ap0 + k0 + hl * 16 + 8);
    a[1].u[0] = *reinterpret_cast<const uint4v*>(Ap1 + k0 + hl * 16);
    a[1].u[1] = *reinterpret_cast<const uint4v*>(Ap1 + k0 + hl * 16 + 8);
    // B 32x16 bf16: elem j -> K = k0 + half*16 + j
    b[0].u[0] = *reinterpret_cast<const uint4v*>(Wp0 + k0 + hl * 16);
    b[0].u[1] = *reinterpret_cast<const uint4v*>(Wp0 + k0 + hl * 16 + 8);
    b[1].u[0] = *reinterpret_cast<const uint4v*>(Wp1 + k0 + hl * 16);
    b[1].u[1] = *reinterpret_cast<const uint4v*>(Wp1 + k0 + hl * 16 + 8);
    b[2].u[0] = *reinterpret_cast<const uint4v*>(Wp2 + k0 + hl * 16);
    b[2].u[1] = *reinterpret_cast<const uint4v*>(Wp2 + k0 + hl * 16 + 8);
    b[3].u[0] = *reinterpret_cast<const uint4v*>(Wp3 + k0 + hl * 16);
    b[3].u[1] = *reinterpret_cast<const uint4v*>(Wp3 + k0 + hl * 16 + 8);
#pragma unroll
    for (int mi = 0; mi < 2; ++mi)
#pragma unroll
      for (int ni = 0; ni < 4; ++ni)
        acc[mi][ni] = __builtin_amdgcn_wmma_f32_16x16x32_bf16(
            false, a[mi].v, false, b[ni].v, (short)0, acc[mi][ni], false, false);
  }

  // C/D layout: VGPR r -> M = mbase + half*8 + r, N = nbase + (lane&15)
#pragma unroll
  for (int mi = 0; mi < 2; ++mi) {
#pragma unroll
    for (int ni = 0; ni < 4; ++ni) {
      int n = n0 + ni * 16 + l16;
      float bn = bias ? bias[n] : 0.0f;
#pragma unroll
      for (int r = 0; r < 8; ++r) {
        int m = m0 + mi * 16 + hl * 8 + r;
        float val = acc[mi][ni][r] + bn;
        if (act == 1) {  // exact GELU
          val = 0.5f * val * (1.0f + erff(val * 0.70710678118654752f));
        }
        int row = scatter ? scatter[m] : m;
        if (outF) outF[(size_t)row * ldo + n] = val;
        else      outB[(size_t)row * ldo + n] = __float2bfloat16(val);
      }
    }
  }
}

// ---------------------------------------------------------------------------
// Attention for one (set, head): scores -> masked softmax -> o, bf16 out.
// qk: (NVOX x 384) bf16 [q | k] in set order; v: (NVOX x 192) bf16.
// All math in fp32 from LDS.
// ---------------------------------------------------------------------------
__global__ __launch_bounds__(128) void attn_kernel(
    const __hip_bfloat16* __restrict__ qk, const __hip_bfloat16* __restrict__ v,
    const unsigned char* __restrict__ mask,
    __hip_bfloat16* __restrict__ o) {
  int s = blockIdx.x;
  int h = blockIdx.y;
  int tid = threadIdx.x;

  __shared__ float qs[KSET * DH];
  __shared__ float ks[KSET * DH];
  __shared__ float vs[KSET * DH];
  __shared__ float sc[KSET * KSET];

  for (int idx = tid; idx < KSET * DH; idx += 128) {
    int i = idx / DH, d = idx % DH;
    size_t rowq = (size_t)(s * KSET + i) * (2 * C_DIM);
    qs[idx] = __bfloat162float(qk[rowq + h * DH + d]);
    ks[idx] = __bfloat162float(qk[rowq + C_DIM + h * DH + d]);
    vs[idx] = __bfloat162float(v[(size_t)(s * KSET + i) * C_DIM + h * DH + d]);
  }
  __syncthreads();

  const float scale = 0.2041241452319315f;  // 1/sqrt(24)
  for (int idx = tid; idx < KSET * KSET; idx += 128) {
    int i = idx / KSET, j = idx % KSET;
    float acc = 0.0f;
#pragma unroll
    for (int d = 0; d < DH; ++d) acc += qs[i * DH + d] * ks[j * DH + d];
    acc *= scale;
    if (mask[(size_t)s * KSET + j]) acc = -1e9f;
    sc[idx] = acc;
  }
  __syncthreads();

  if (tid < KSET) {
    float mx = -3.4e38f;
    for (int j = 0; j < KSET; ++j) mx = fmaxf(mx, sc[tid * KSET + j]);
    float sum = 0.0f;
    for (int j = 0; j < KSET; ++j) {
      float e = __expf(sc[tid * KSET + j] - mx);
      sc[tid * KSET + j] = e;
      sum += e;
    }
    float inv = 1.0f / sum;
    for (int j = 0; j < KSET; ++j) sc[tid * KSET + j] *= inv;
  }
  __syncthreads();

  for (int idx = tid; idx < KSET * DH; idx += 128) {
    int i = idx / DH, d = idx % DH;
    float acc = 0.0f;
#pragma unroll
    for (int j = 0; j < KSET; ++j) acc += sc[i * KSET + j] * vs[j * DH + d];
    o[(size_t)(s * KSET + i) * C_DIM + h * DH + d] = __float2bfloat16(acc);
  }
}

// ---------------------------------------------------------------------------
// out = LayerNorm(xin + res) * g + b   (row-wise over C=192). One wave/row.
// Optionally also emits bf16 copy (feeds the next GEMM's A matrix).
// ---------------------------------------------------------------------------
__global__ __launch_bounds__(256) void ln_residual_kernel(
    const float* __restrict__ xin, const float* __restrict__ res,
    const float* __restrict__ g, const float* __restrict__ b,
    float* __restrict__ outF, __hip_bfloat16* __restrict__ outB, int nrows) {
  int wid = threadIdx.x >> 5, lane = threadIdx.x & 31;
  int row = blockIdx.x * 8 + wid;
  if (row >= nrows) return;
  size_t base = (size_t)row * C_DIM;
  float vals[6];
  float s = 0.0f;
#pragma unroll
  for (int t = 0; t < 6; ++t) {
    int c = t * 32 + lane;
    float xv = xin[base + c] + res[base + c];
    vals[t] = xv;
    s += xv;
  }
#pragma unroll
  for (int off = 16; off >= 1; off >>= 1) s += __shfl_xor(s, off, 32);
  float mean = s * (1.0f / C_DIM);
  float v2 = 0.0f;
#pragma unroll
  for (int t = 0; t < 6; ++t) { float d = vals[t] - mean; v2 += d * d; }
#pragma unroll
  for (int off = 16; off >= 1; off >>= 1) v2 += __shfl_xor(v2, off, 32);
  float rstd = rsqrtf(v2 * (1.0f / C_DIM) + 1e-5f);
#pragma unroll
  for (int t = 0; t < 6; ++t) {
    int c = t * 32 + lane;
    float y = (vals[t] - mean) * rstd * g[c] + b[c];
    outF[base + c] = y;
    if (outB) outB[base + c] = __float2bfloat16(y);
  }
}

// ---------------------------------------------------------------------------
// Host orchestration
// ---------------------------------------------------------------------------
extern "C" void kernel_launch(void* const* d_in, const int* in_sizes, int n_in,
                              void* d_out, int out_size, void* d_ws, size_t ws_size,
                              hipStream_t stream) {
  (void)in_sizes; (void)n_in; (void)out_size; (void)ws_size;

  const float* src        = (const float*)d_in[0];
  const float* pos_embed  = (const float*)d_in[1];
  const int*   sv_inds    = (const int*)d_in[2];
  const unsigned char* sv_masks = (const unsigned char*)d_in[3];
  const float* in_proj_w  = (const float*)d_in[4];
  const float* in_proj_b  = (const float*)d_in[5];
  const float* out_w      = (const float*)d_in[6];
  const float* out_b      = (const float*)d_in[7];
  const float* lin1_w     = (const float*)d_in[8];
  const float* lin1_b     = (const float*)d_in[9];
  const float* lin2_w     = (const float*)d_in[10];
  const float* lin2_b     = (const float*)d_in[11];
  const float* ln1_g      = (const float*)d_in[12];
  const float* ln1_b      = (const float*)d_in[13];
  const float* ln2_g      = (const float*)d_in[14];
  const float* ln2_b      = (const float*)d_in[15];
  const float* enc_g      = (const float*)d_in[16];
  const float* enc_b      = (const float*)d_in[17];
  const float* blk_g      = (const float*)d_in[18];
  const float* blk_b      = (const float*)d_in[19];

  // ---- workspace carve-out (256B aligned slabs) ----
  size_t off = 0;
  auto take = [&](size_t bytes) -> void* {
    void* p = (char*)d_ws + off;
    off += (bytes + 255) & ~(size_t)255;
    return p;
  };
  __hip_bfloat16* wqkv_bf = (__hip_bfloat16*)take((size_t)LAYERS * 3 * C_DIM * C_DIM * 2);
  __hip_bfloat16* wout_bf = (__hip_bfloat16*)take((size_t)LAYERS * C_DIM * C_DIM * 2);
  __hip_bfloat16* w1_bf   = (__hip_bfloat16*)take((size_t)LAYERS * FDIM * C_DIM * 2);
  __hip_bfloat16* w2_bf   = (__hip_bfloat16*)take((size_t)LAYERS * C_DIM * FDIM * 2);
  __hip_bfloat16* bufA    = (__hip_bfloat16*)take((size_t)NVOX * FDIM * 2);   // qk_in / h1
  __hip_bfloat16* bufB    = (__hip_bfloat16*)take((size_t)NVOX * C_DIM * 2);  // g / o_bf / x_bf
  __hip_bfloat16* qk_bf   = (__hip_bfloat16*)take((size_t)NVOX * 2 * C_DIM * 2);
  __hip_bfloat16* v_bf    = (__hip_bfloat16*)take((size_t)NVOX * C_DIM * 2);
  float* src2 = (float*)take((size_t)NVOX * C_DIM * 4);
  float* pool[4];
  for (int t = 0; t < 4; ++t) pool[t] = (float*)take((size_t)NVOX * C_DIM * 4);

  // pick a rotation buffer not aliasing any live pointer
  auto pick = [&](const float* a0, const float* a1, const float* a2) -> float* {
    for (int t = 0; t < 4; ++t)
      if (pool[t] != a0 && pool[t] != a1 && pool[t] != a2) return pool[t];
    return pool[0];
  };

  // ---- weight conversion fp32 -> bf16 ----
  auto cvt = [&](const float* in, __hip_bfloat16* out, size_t n) {
    f32_to_bf16_kernel<<<(unsigned)((n + 255) / 256), 256, 0, stream>>>(in, out, n);
  };
  cvt(in_proj_w, wqkv_bf, (size_t)LAYERS * 3 * C_DIM * C_DIM);
  cvt(out_w,     wout_bf, (size_t)LAYERS * C_DIM * C_DIM);
  cvt(lin1_w,    w1_bf,   (size_t)LAYERS * FDIM * C_DIM);
  cvt(lin2_w,    w2_bf,   (size_t)LAYERS * C_DIM * FDIM);

  auto gemm = [&](const __hip_bfloat16* A, int lda, const __hip_bfloat16* W, int ldw,
                  const float* bias, int M, int Ncols, int Kd, const int* scatter,
                  float* outF, __hip_bfloat16* outB, int ldo, int act) {
    int tiles = (M >> 5) * (Ncols >> 6);
    int blocks = (tiles + 7) / 8;
    gemm_bf16_wmma_kernel<<<blocks, 256, 0, stream>>>(
        A, lda, W, ldw, bias, M, Ncols, Kd, scatter, outF, outB, ldo, act);
  };
  auto ln = [&](const float* xx, const float* res, const float* gg, const float* bb,
                float* outF, __hip_bfloat16* outB) {
    ln_residual_kernel<<<(NVOX + 7) / 8, 256, 0, stream>>>(xx, res, gg, bb, outF, outB, NVOX);
  };

  const unsigned elem_blocks = (unsigned)(((size_t)NVOX * C_DIM + 255) / 256);

  const float* cur = src;  // running activation (read-only view)
  for (int blk = 0; blk < NBLOCKS; ++blk) {
    int shift = blk % 2;
    const float* resid = cur;  // block residual: pointer snapshot, no copy

    for (int i = 0; i < 2; ++i) {
      int li = blk * 2 + i;
      const int* inds = sv_inds + ((size_t)(shift * 2 + i)) * SETS * KSET;
      const unsigned char* mask = sv_masks + ((size_t)(shift * 2 + i)) * SETS * KSET;
      const float* pos = pos_embed + (size_t)i * NVOX * C_DIM;
      const float* identity = cur;

      // gather + pos add, emit bf16 A matrices
      gather_prep_kernel<<<elem_blocks, 256, 0, stream>>>(cur, pos, inds, bufA, bufB);

      // [q|k] = qk_in @ [Wq;Wk]^T + b   (M x 384, bf16 out)
      gemm(bufA, C_DIM, wqkv_bf + (size_t)li * 3 * C_DIM * C_DIM, C_DIM,
           in_proj_b + (size_t)li * 3 * C_DIM,
           NVOX, 2 * C_DIM, C_DIM, nullptr, nullptr, qk_bf, 2 * C_DIM, 0);
      // v = g @ Wv^T + bv               (M x 192, bf16 out)
      gemm(bufB, C_DIM, wqkv_bf + (size_t)li * 3 * C_DIM * C_DIM + (size_t)2 * C_DIM * C_DIM,
           C_DIM, in_proj_b + (size_t)li * 3 * C_DIM + 2 * C_DIM,
           NVOX, C_DIM, C_DIM, nullptr, nullptr, v_bf, C_DIM, 0);

      // per-(set, head) softmax attention -> o (bf16, set order) in bufB
      attn_kernel<<<dim3(SETS, H_DIM), 128, 0, stream>>>(qk_bf, v_bf, mask, bufB);

      // out-proj with fused inverse-permutation scatter -> src2 (voxel order)
      gemm(bufB, C_DIM, wout_bf + (size_t)li * C_DIM * C_DIM, C_DIM,
           out_b + (size_t)li * C_DIM,
           NVOX, C_DIM, C_DIM, inds, src2, nullptr, C_DIM, 0);

      // x1 = LN(x + src2); also emit bf16 for FFN GEMM A
      float* x1 = pick(cur, resid, nullptr);
      ln(cur, src2, ln1_g + (size_t)li * C_DIM, ln1_b + (size_t)li * C_DIM, x1, bufB);

      // FFN: h1 = GELU(x1 @ W1^T + b1) (bf16) ; ff = h1 @ W2^T + b2 (fp32)
      gemm(bufB, C_DIM, w1_bf + (size_t)li * FDIM * C_DIM, C_DIM,
           lin1_b + (size_t)li * FDIM,
           NVOX, FDIM, C_DIM, nullptr, nullptr, bufA, FDIM, 1);
      gemm(bufA, FDIM, w2_bf + (size_t)li * C_DIM * FDIM, FDIM,
           lin2_b + (size_t)li * C_DIM,
           NVOX, C_DIM, FDIM, nullptr, src2, nullptr, C_DIM, 0);

      // x2 = LN(x1 + ff); x3 = LN(x2 + identity)
      float* x2 = pick(x1, cur, resid);
      ln(x1, src2, ln2_g + (size_t)li * C_DIM, ln2_b + (size_t)li * C_DIM, x2, nullptr);
      float* x3 = pick(x2, cur, resid);
      ln(x2, identity, enc_g + (size_t)li * C_DIM, enc_b + (size_t)li * C_DIM, x3, nullptr);
      cur = x3;
    }

    // block-level LN with residual; final block writes straight to d_out
    float* xb = (blk == NBLOCKS - 1) ? (float*)d_out : pick(cur, resid, nullptr);
    ln(cur, resid, blk_g + (size_t)blk * C_DIM, blk_b + (size_t)blk * C_DIM, xb, nullptr);
    cur = xb;
  }
}
